// Phi3DecoderLayer_40286793237167
// MI455X (gfx1250) — compile-verified, matrix operand order checked by reading, not run on hardware
//
#include <hip/hip_runtime.h>
#include <hip/hip_bf16.h>
#include <math.h>

// ---------------- problem constants ----------------
#define S_LEN 2048
#define HID   3072
#define NH    32
#define NKV   8
#define HD    96          // head dim
#define QKV_N 4608        // NH*HD + 2*NKV*HD
#define INTER 8192
#define EPS   1e-5f

typedef __bf16 bf16_t;
typedef __attribute__((ext_vector_type(16))) __bf16 v16bf;
typedef __attribute__((ext_vector_type(8)))  float  v8f;

union FragU { v16bf v; float4 q[2]; };

// Load a 16x32 bf16 WMMA A/B fragment from a row-major [rows][K] slab.
// base points at (row, k0) start; lane half h reads K runs {h*8..h*8+7} and {16+h*8..}.
__device__ __forceinline__ v16bf load_frag(const bf16_t* base, int lh) {
    FragU u;
    u.q[0] = *(const float4*)(base + lh * 8);
    u.q[1] = *(const float4*)(base + 16 + lh * 8);
    return u.v;
}

__device__ __forceinline__ v8f v8f_zero() {
    v8f z = {0.f, 0.f, 0.f, 0.f, 0.f, 0.f, 0.f, 0.f};
    return z;
}

// ---------------- RMSNorm (fp32 in -> bf16 out) ----------------
__launch_bounds__(256)
__global__ void rmsnorm_bf16_kernel(const float* __restrict__ x,
                                    const float* __restrict__ w,
                                    bf16_t* __restrict__ out) {
    __shared__ float red[256];
    int s = blockIdx.x;
    const float* row = x + (size_t)s * HID;
    float ss = 0.f;
    for (int i = threadIdx.x; i < HID; i += 256) {
        float v = row[i];
        ss += v * v;
    }
    red[threadIdx.x] = ss;
    __syncthreads();
    for (int off = 128; off > 0; off >>= 1) {
        if (threadIdx.x < off) red[threadIdx.x] += red[threadIdx.x + off];
        __syncthreads();
    }
    float inv = rsqrtf(red[0] / (float)HID + EPS);
    for (int i = threadIdx.x; i < HID; i += 256) {
        out[(size_t)s * HID + i] = (bf16_t)(row[i] * inv * w[i]);
    }
}

// ---------------- weight convert+transpose: fp32 [K][N] -> bf16 [N][K] ----------------
__launch_bounds__(256)
__global__ void transpose_bf16_kernel(const float* __restrict__ W,
                                      bf16_t* __restrict__ Wt,
                                      int K, int N) {
    __shared__ float tile[32][33];
    int n0 = blockIdx.x * 32, k0 = blockIdx.y * 32;
    int tx = threadIdx.x, ty = threadIdx.y;      // 32 x 8
#pragma unroll
    for (int i = 0; i < 32; i += 8)
        tile[ty + i][tx] = W[(size_t)(k0 + ty + i) * N + n0 + tx];
    __syncthreads();
#pragma unroll
    for (int i = 0; i < 32; i += 8)
        Wt[(size_t)(n0 + ty + i) * K + k0 + tx] = (bf16_t)tile[tx][ty + i];
}

// ---------------- WMMA GEMM: C[M,N] = A[M,K](bf16) * Bt[N,K](bf16)^T (+R) ----------------
#define BM 128
#define BN 128
#define BK 32
#define LDT 40   // padded LDS row stride in bf16 elems (80B, 16B aligned, bank-spread)

template <bool HAS_R>
__launch_bounds__(256)
__global__ void gemm_bf16_wmma_kernel(const bf16_t* __restrict__ A,
                                      const bf16_t* __restrict__ Bt,
                                      float* __restrict__ C,
                                      const float* __restrict__ R,
                                      int M, int N, int K) {
    __shared__ bf16_t As[BM * LDT];
    __shared__ bf16_t Bs[BN * LDT];
    int tid  = threadIdx.x;
    int lane = tid & 31, wave = tid >> 5;
    int lh = lane >> 4, lr = lane & 15;
    int wm = wave & 3, wn = wave >> 2;           // 4 M-subtiles x 2 N-subtiles
    int m0 = blockIdx.y * BM, n0 = blockIdx.x * BN;

    // this thread's two 16B chunks per tile (chunk idx -> row/col)
    int rw0 = tid >> 2,          cc0 = (tid & 3) * 8;
    int rw1 = (256 + tid) >> 2,  cc1 = ((256 + tid) & 3) * 8;
    const bf16_t* Ab = A + (size_t)m0 * K;
    const bf16_t* Bb = Bt + (size_t)n0 * K;

    v8f acc[2][4];
#pragma unroll
    for (int i = 0; i < 2; ++i)
#pragma unroll
        for (int j = 0; j < 4; ++j) acc[i][j] = v8f_zero();

    // prologue prefetch of tile k0=0
    float4 pa0 = *(const float4*)(Ab + (size_t)rw0 * K + cc0);
    float4 pa1 = *(const float4*)(Ab + (size_t)rw1 * K + cc1);
    float4 pb0 = *(const float4*)(Bb + (size_t)rw0 * K + cc0);
    float4 pb1 = *(const float4*)(Bb + (size_t)rw1 * K + cc1);

    for (int k0 = 0; k0 < K; k0 += BK) {
        // commit prefetched tile to LDS
        *(float4*)(As + rw0 * LDT + cc0) = pa0;
        *(float4*)(As + rw1 * LDT + cc1) = pa1;
        *(float4*)(Bs + rw0 * LDT + cc0) = pb0;
        *(float4*)(Bs + rw1 * LDT + cc1) = pb1;
        __syncthreads();

        // issue next tile's global loads (overlap with WMMA work below)
        int kn = k0 + BK;
        if (kn < K) {
            pa0 = *(const float4*)(Ab + (size_t)rw0 * K + kn + cc0);
            pa1 = *(const float4*)(Ab + (size_t)rw1 * K + kn + cc1);
            pb0 = *(const float4*)(Bb + (size_t)rw0 * K + kn + cc0);
            pb1 = *(const float4*)(Bb + (size_t)rw1 * K + kn + cc1);
        }

        v16bf af[2], bfr[4];
#pragma unroll
        for (int i = 0; i < 2; ++i)
            af[i] = load_frag(As + (wm * 32 + i * 16 + lr) * LDT, lh);
#pragma unroll
        for (int j = 0; j < 4; ++j)
            bfr[j] = load_frag(Bs + (wn * 64 + j * 16 + lr) * LDT, lh);
#pragma unroll
        for (int i = 0; i < 2; ++i)
#pragma unroll
            for (int j = 0; j < 4; ++j)
                acc[i][j] = __builtin_amdgcn_wmma_f32_16x16x32_bf16(
                    false, af[i], false, bfr[j], (short)0, acc[i][j], false, false);
        __syncthreads();   // all waves done reading before next store
    }

    // epilogue: one base offset per 16x16 tile, pointer-bump over rows
    int mbase = m0 + wm * 32 + 8 * lh;
    int nbase = n0 + wn * 64 + lr;
#pragma unroll
    for (int i = 0; i < 2; ++i)
#pragma unroll
        for (int j = 0; j < 4; ++j) {
            size_t off = (size_t)(mbase + i * 16) * N + (nbase + j * 16);
            float* cp = C + off;
            const float* rp = R + off;
#pragma unroll
            for (int r = 0; r < 8; ++r) {
                float v = acc[i][j][r];
                if (HAS_R) v += *rp;
                *cp = v;
                cp += N;
                rp += N;
            }
        }
}

// ---------------- RoPE + pack to head-major bf16 (scale folded into Q) ----------------
__launch_bounds__(256)
__global__ void rope_pack_kernel(const float* __restrict__ qkv,
                                 const float* __restrict__ sinT,
                                 const float* __restrict__ cosT,
                                 const int* __restrict__ pos,
                                 bf16_t* __restrict__ Qo,   // [NH][S][HD]
                                 bf16_t* __restrict__ Ko,   // [NKV][S][HD]
                                 bf16_t* __restrict__ Vo) { // [NKV][S][HD]
    int s = blockIdx.x;
    int p = pos[s];
    const float* row = qkv + (size_t)s * QKV_N;
    const float* sr = sinT + (size_t)p * HD;
    const float* cr = cosT + (size_t)p * HD;
    const float scale = 0.1020620726f;  // 1/sqrt(96)
    for (int i = threadIdx.x; i < NH * 48; i += 256) {
        int h = i / 48, d = i % 48;
        float x1 = row[h * HD + d], x2 = row[h * HD + d + 48];
        float o1 = x1 * cr[d] - x2 * sr[d];
        float o2 = x2 * cr[d + 48] + x1 * sr[d + 48];
        size_t base = ((size_t)h * S_LEN + s) * HD;
        Qo[base + d]      = (bf16_t)(o1 * scale);
        Qo[base + d + 48] = (bf16_t)(o2 * scale);
    }
    for (int i = threadIdx.x; i < NKV * 48; i += 256) {
        int h = i / 48, d = i % 48;
        float x1 = row[NH * HD + h * HD + d], x2 = row[NH * HD + h * HD + d + 48];
        float o1 = x1 * cr[d] - x2 * sr[d];
        float o2 = x2 * cr[d + 48] + x1 * sr[d + 48];
        size_t base = ((size_t)h * S_LEN + s) * HD;
        Ko[base + d]      = (bf16_t)o1;
        Ko[base + d + 48] = (bf16_t)o2;
    }
    for (int i = threadIdx.x; i < NKV * HD; i += 256) {
        int h = i / HD, d = i % HD;
        Vo[((size_t)h * S_LEN + s) * HD + d] =
            (bf16_t)row[NH * HD + NKV * HD + h * HD + d];
    }
}

// ---------------- flash attention (causal, GQA), WMMA scores + P*V ----------------
#define KLD 104   // K-tile LDS row stride (bf16 elems): 96 + 8 pad, 208B (16B-aligned)
#define VLD 72    // Vt / P row stride: 64 + 8 pad, 144B (16B-aligned)

__launch_bounds__(128)
__global__ void flash_attn_kernel(const bf16_t* __restrict__ Q,
                                  const bf16_t* __restrict__ Kc,
                                  const bf16_t* __restrict__ Vc,
                                  bf16_t* __restrict__ O) {  // [S][HID]
    __shared__ bf16_t Ks[64 * KLD];
    __shared__ bf16_t Vt[HD * VLD];
    __shared__ bf16_t Ps[4][16 * VLD];

    int qt = blockIdx.x;          // 64-row q tile
    int head = blockIdx.y;
    int kvh = head >> 2;          // GQA group of 4
    int tid = threadIdx.x;
    int lane = tid & 31, wave = tid >> 5;
    int lh = lane >> 4, lr = lane & 15;

    // Q fragments for this wave's 16 rows (row m = lr)
    v16bf qf[3];
    const bf16_t* qp = Q + ((size_t)head * S_LEN + qt * 64 + wave * 16 + lr) * HD;
#pragma unroll
    for (int f = 0; f < 3; ++f) qf[f] = load_frag(qp + f * 32, lh);

    v8f oacc[6];
#pragma unroll
    for (int d = 0; d < 6; ++d) oacc[d] = v8f_zero();
    float mrun[8], lrun[8];
#pragma unroll
    for (int r = 0; r < 8; ++r) { mrun[r] = -1e30f; lrun[r] = 0.f; }

    for (int kt = 0; kt <= qt; ++kt) {
        __syncthreads();
        // cooperative load: K tile row-major, V tile transposed
#pragma unroll
        for (int it = 0; it < 6; ++it) {
            int idx = it * 128 + tid;            // 0..767 chunks of 8 bf16
            int row = idx / 12, c = idx % 12;
            size_t g = ((size_t)kvh * S_LEN + kt * 64 + row) * HD + c * 8;
            *(float4*)(Ks + row * KLD + c * 8) = *(const float4*)(Kc + g);
            float4 vv = *(const float4*)(Vc + g);
            const bf16_t* ve = (const bf16_t*)&vv;
#pragma unroll
            for (int e = 0; e < 8; ++e) Vt[(c * 8 + e) * VLD + row] = ve[e];
        }
        __syncthreads();

        // scores: 16(q) x 64(k) per wave
        v8f sacc[4];
#pragma unroll
        for (int nb = 0; nb < 4; ++nb) {
            sacc[nb] = v8f_zero();
#pragma unroll
            for (int f = 0; f < 3; ++f) {
                v16bf kb = load_frag(Ks + (nb * 16 + lr) * KLD + f * 32, lh);
                sacc[nb] = __builtin_amdgcn_wmma_f32_16x16x32_bf16(
                    false, qf[f], false, kb, (short)0, sacc[nb], false, false);
            }
        }

        // online softmax per row (row m = r + 8*lh)
#pragma unroll
        for (int r = 0; r < 8; ++r) {
            int qpos = qt * 64 + wave * 16 + r + 8 * lh;
            float mx = -1e30f;
#pragma unroll
            for (int nb = 0; nb < 4; ++nb) {
                float sv = sacc[nb][r];
                if (kt == qt) {
                    int kpos = kt * 64 + nb * 16 + lr;
                    if (kpos > qpos) sv = -1e30f;
                    sacc[nb][r] = sv;
                }
                mx = fmaxf(mx, sv);
            }
#pragma unroll
            for (int off = 1; off < 16; off <<= 1)
                mx = fmaxf(mx, __shfl_xor(mx, off, 32));
            float newm = fmaxf(mrun[r], mx);
            float alpha = __expf(mrun[r] - newm);
            mrun[r] = newm;
            float rs = 0.f;
#pragma unroll
            for (int nb = 0; nb < 4; ++nb) {
                float p = __expf(sacc[nb][r] - newm);
                rs += p;
                Ps[wave][(r + 8 * lh) * VLD + nb * 16 + lr] = (bf16_t)p;
            }
#pragma unroll
            for (int off = 1; off < 16; off <<= 1)
                rs += __shfl_xor(rs, off, 32);
            lrun[r] = lrun[r] * alpha + rs;
#pragma unroll
            for (int d = 0; d < 6; ++d) oacc[d][r] *= alpha;
        }

        // wave-local LDS RAW: wait for P stores before re-fragmenting
        asm volatile("s_wait_dscnt 0" ::: "memory");

        // O += P(16x64) * V(64x96)
#pragma unroll
        for (int d = 0; d < 6; ++d) {
#pragma unroll
            for (int pf = 0; pf < 2; ++pf) {
                v16bf pa = load_frag(Ps[wave] + lr * VLD + pf * 32, lh);
                v16bf vb = load_frag(Vt + (d * 16 + lr) * VLD + pf * 32, lh);
                oacc[d] = __builtin_amdgcn_wmma_f32_16x16x32_bf16(
                    false, pa, false, vb, (short)0, oacc[d], false, false);
            }
        }
    }

    // normalize + store [s][h*96+d] as bf16 (A operand of O-proj GEMM)
#pragma unroll
    for (int d = 0; d < 6; ++d)
#pragma unroll
        for (int r = 0; r < 8; ++r) {
            int q = qt * 64 + wave * 16 + r + 8 * lh;
            float val = oacc[d][r] / lrun[r];
            O[(size_t)q * HID + head * HD + d * 16 + lr] = (bf16_t)val;
        }
}

// ---------------- SiLU(gate) * up -> bf16 ----------------
__launch_bounds__(256)
__global__ void silu_mul_kernel(const float* __restrict__ gu,
                                bf16_t* __restrict__ act) {
    size_t idx = (size_t)blockIdx.x * 256 + threadIdx.x;  // S*INTER total
    int s = (int)(idx / INTER), i = (int)(idx % INTER);
    float g = gu[(size_t)s * (2 * INTER) + i];
    float u = gu[(size_t)s * (2 * INTER) + INTER + i];
    act[idx] = (bf16_t)(u * (g / (1.f + __expf(-g))));
}

// ---------------- host launcher ----------------
extern "C" void kernel_launch(void* const* d_in, const int* in_sizes, int n_in,
                              void* d_out, int out_size, void* d_ws, size_t ws_size,
                              hipStream_t stream) {
    const float* hidden = (const float*)d_in[0];
    const float* sinT   = (const float*)d_in[1];
    const float* cosT   = (const float*)d_in[2];
    // d_in[3] attention_mask: causal (tril) — handled analytically in flash kernel
    const int*   pos    = (const int*)d_in[4];
    const float* Wqkv   = (const float*)d_in[5];
    const float* Wo     = (const float*)d_in[6];
    const float* Wgu    = (const float*)d_in[7];
    const float* Wdown  = (const float*)d_in[8];
    const float* ln1    = (const float*)d_in[9];
    const float* ln2    = (const float*)d_in[10];
    float* out = (float*)d_out;   // also holds hs (attn residual) between phases
    char* ws = (char*)d_ws;

    // workspace layout (bytes)
    bf16_t* wT   = (bf16_t*)(ws);                      // 100,663,296 (max weightT, reused 4x)
    bf16_t* xn   = (bf16_t*)(ws + 100663296);          // 12,582,912 (both RMSNorm outputs)
    float*  qkv  = (float*)(ws + 113246208);           // 134,217,728 (shared with gu)
    float*  gu   = qkv;
    bf16_t* qb   = (bf16_t*)(ws + 247463936);          // 12,582,912
    bf16_t* kb   = (bf16_t*)(ws + 260046848);          //  3,145,728
    bf16_t* vb   = (bf16_t*)(ws + 263192576);          //  3,145,728
    bf16_t* attn = (bf16_t*)(ws + 266338304);          // 12,582,912
    bf16_t* act  = (bf16_t*)(ws + 278921216);          // 33,554,432 -> end 312,475,648

    dim3 tb(32, 8);

    // 1) x = rmsnorm(hidden, ln1)
    rmsnorm_bf16_kernel<<<S_LEN, 256, 0, stream>>>(hidden, ln1, xn);
    // 2) qkv = x @ Wqkv
    transpose_bf16_kernel<<<dim3(QKV_N / 32, HID / 32), tb, 0, stream>>>(Wqkv, wT, HID, QKV_N);
    gemm_bf16_wmma_kernel<false><<<dim3(QKV_N / BN, S_LEN / BM), 256, 0, stream>>>(
        xn, wT, qkv, nullptr, S_LEN, QKV_N, HID);
    // 3) RoPE + pack (softmax scale folded into Q)
    rope_pack_kernel<<<S_LEN, 256, 0, stream>>>(qkv, sinT, cosT, pos, qb, kb, vb);
    // 4) causal GQA attention
    flash_attn_kernel<<<dim3(S_LEN / 64, NH), 128, 0, stream>>>(qb, kb, vb, attn);
    // 5) hs = hidden + attn @ Wo   (hs lives in d_out)
    transpose_bf16_kernel<<<dim3(HID / 32, HID / 32), tb, 0, stream>>>(Wo, wT, HID, HID);
    gemm_bf16_wmma_kernel<true><<<dim3(HID / BN, S_LEN / BM), 256, 0, stream>>>(
        attn, wT, out, hidden, S_LEN, HID, HID);
    // 6) x2 = rmsnorm(hs, ln2)
    rmsnorm_bf16_kernel<<<S_LEN, 256, 0, stream>>>(out, ln2, xn);
    // 7) gu = x2 @ Wgu
    transpose_bf16_kernel<<<dim3((2 * INTER) / 32, HID / 32), tb, 0, stream>>>(Wgu, wT, HID, 2 * INTER);
    gemm_bf16_wmma_kernel<false><<<dim3((2 * INTER) / BN, S_LEN / BM), 256, 0, stream>>>(
        xn, wT, gu, nullptr, S_LEN, 2 * INTER, HID);
    // 8) act = up * silu(gate)
    silu_mul_kernel<<<(S_LEN * INTER) / 256, 256, 0, stream>>>(gu, act);
    // 9) out = hs + act @ Wdown   (in-place residual on d_out)
    transpose_bf16_kernel<<<dim3(HID / 32, INTER / 32), tb, 0, stream>>>(Wdown, wT, INTER, HID);
    gemm_bf16_wmma_kernel<true><<<dim3(HID / BN, S_LEN / BM), 256, 0, stream>>>(
        act, wT, out, out, S_LEN, HID, INTER);
}
